// SimpleMamba_16303695855798
// MI455X (gfx1250) — matrix operand order repaired and checked by reading, not compiled, not run on hardware
//
#include <hip/hip_runtime.h>
#include <hip/hip_bf16.h>
#include <math.h>

// ---- problem sizes (compile-time) ----
#define I_DIM 256
#define H_DIM 512
#define O_DIM 256
#define L_DIM 2
#define B_DIM 16
#define S_DIM 2048
#define M_DIM (B_DIM * S_DIM)   // 32768
#define EPS_LN 1e-5f

#define USE_TDM 1

typedef __bf16 bf16;
typedef __attribute__((ext_vector_type(16))) __bf16 v16bf;
typedef __attribute__((ext_vector_type(8)))  float  v8f;
typedef __attribute__((ext_vector_type(4)))  unsigned int v4u;
typedef __attribute__((ext_vector_type(8)))  int  v8i;
typedef __attribute__((ext_vector_type(4)))  int  v4i;

struct bf16x16 { uint4 lo, hi; };

// Load a 16-element bf16 fragment from two 16-byte chunks (matches CDNA5
// 16-bit A/B WMMA VGPR layout: lane holds two 8-element K-runs).
static __device__ __forceinline__ v16bf ld_frag(const bf16* p0, const bf16* p1) {
  bf16x16 t;
  t.lo = *(const uint4*)p0;
  t.hi = *(const uint4*)p1;
  return __builtin_bit_cast(v16bf, t);
}

#if USE_TDM
// Issue a TDM 2-D tile load (global -> LDS), 2-byte elements, with LDS row
// padding per D# pad fields (ISA 08_async_tensor §8.3/8.4). Uniform args only;
// call from a single wave (TDM ignores EXEC — one issue per executing wave).
// This toolchain exposes the 6-arg builtin:
//   (uint32x4 g0, int32x8 g1, int32x4 g2, int32x4 g3, int32x8 ext, i32 cpol)
static __device__ __forceinline__ void tdm_load_tile_2d(
    unsigned lds_byte_addr, const void* gptr,
    unsigned tensor_d0, unsigned tensor_d1,   // tensor extent (elements)
    unsigned tile_d0, unsigned tile_d1,       // tile extent (elements / rows)
    unsigned stride0_elems,                   // row stride (elements)
    unsigned pad_interval_enc, unsigned pad_amount_enc)
{
  unsigned long long ga = (unsigned long long)(uintptr_t)gptr;
  v4u g0;
  g0.x = 1u;                                            // count=1, user mode
  g0.y = lds_byte_addr;                                 // lds_addr [63:32]
  g0.z = (unsigned)(ga & 0xFFFFFFFFull);                // global_addr lo
  g0.w = (unsigned)((ga >> 32) & 0x01FFFFFFull)         // global_addr hi [120:96]
       | (2u << 30);                                    // type=2 ("image")
  v8i g1;
  g1[0] = (int)((1u << 16)                              // data_size=1 -> 2 bytes
              | (1u << 20)                              // pad_enable
              | (pad_interval_enc << 22)                // pad_interval [24:22]
              | (pad_amount_enc << 25));                // pad_amount   [31:25]
  g1[1] = (int)((tensor_d0 & 0xFFFFu) << 16);           // tensor_dim0 [79:48]
  g1[2] = (int)(((tensor_d0 >> 16) & 0xFFFFu)
              | ((tensor_d1 & 0xFFFFu) << 16));         // tensor_dim1 [111:80]
  g1[3] = (int)(((tensor_d1 >> 16) & 0xFFFFu)
              | (tile_d0 << 16));                       // tile_dim0 [127:112]
  g1[4] = (int)(tile_d1 & 0xFFFFu);                     // tile_dim1; tile_dim2=0
  g1[5] = (int)stride0_elems;                           // tensor_dim0_stride lo
  g1[6] = 0;                                            // stride0 hi / stride1 lo
  g1[7] = 0;
  v4i z4 = {0, 0, 0, 0};                                // 2-D: groups 2/3 unused
  v8i z8 = {0, 0, 0, 0, 0, 0, 0, 0};
  __builtin_amdgcn_tensor_load_to_lds(g0, g1, z4, z4, z8, 0);
}
#endif

// ---------------- f32 -> bf16 convert ----------------
__global__ void k_cvt(const float* __restrict__ src, bf16* __restrict__ dst, int n) {
  int i = blockIdx.x * blockDim.x + threadIdx.x;
  int stride = gridDim.x * blockDim.x;
  for (; i < n; i += stride) dst[i] = (bf16)src[i];
}

// ---------------- WMMA GEMM: out[M,N] = A1[M,K1]*W1[N,K1]^T (+ A2*W2^T) + bias ----------------
#define BM 64
#define BN 128
#define KC 64
#define LDT 72   // KC + 8 bf16 pad -> 144B row (16B aligned, bank-staggered)
                 // TDM pad: 32 DW data (pad_interval enc 4) + 4 DW pad (enc 3)
#define PAD_IVL_ENC 4u
#define PAD_AMT_ENC 3u

template <bool DUAL, bool OUTB>
__global__ __launch_bounds__(256) void k_gemm(
    const bf16* __restrict__ A1, const bf16* __restrict__ W1,
    const float* __restrict__ bias1, int K1,
    const bf16* __restrict__ A2, const bf16* __restrict__ W2,
    const float* __restrict__ bias2, int K2,
    float* __restrict__ outF, bf16* __restrict__ outB, int N, int Mtot)
{
  __shared__ __align__(16) bf16 As[2][BM][LDT];
  __shared__ __align__(16) bf16 Ws[2][BN][LDT];

  const int tid  = threadIdx.x;
  const int lane = tid & 31;
  const int wave = tid >> 5;
  const int wm = wave >> 2, wn = wave & 3;     // 2 x 4 wave grid -> 32x32 per wave
  const int m0 = blockIdx.y * BM;
  const int n0 = blockIdx.x * BN;
  const int l16 = lane & 15, hlf = lane >> 4;

  const v8f vzero = {};
  v8f acc[2][2];
  #pragma unroll
  for (int i = 0; i < 2; ++i)
    #pragma unroll
    for (int j = 0; j < 2; ++j) acc[i][j] = vzero;

  const int npass = DUAL ? 2 : 1;

  #pragma unroll 1
  for (int pass = 0; pass < npass; ++pass) {
    const bf16* A = (DUAL && pass) ? A2 : A1;
    const bf16* W = (DUAL && pass) ? W2 : W1;
    const int   K = (DUAL && pass) ? K2 : K1;

#if USE_TDM
    // prologue: DMA first tile pair into buffer 0 (wave 0 issues; TENSORcnt)
    if (wave == 0) {
      tdm_load_tile_2d((unsigned)(uintptr_t)&As[0][0][0], &A[(size_t)m0 * K],
                       (unsigned)K, (unsigned)Mtot, KC, BM, (unsigned)K,
                       PAD_IVL_ENC, PAD_AMT_ENC);
      tdm_load_tile_2d((unsigned)(uintptr_t)&Ws[0][0][0], &W[(size_t)n0 * K],
                       (unsigned)K, (unsigned)N, KC, BN, (unsigned)K,
                       PAD_IVL_ENC, PAD_AMT_ENC);
    }
    int nb = 0;
    #pragma unroll 1
    for (int kc = 0; kc < K; kc += KC) {
      if (wave == 0) __builtin_amdgcn_s_wait_tensorcnt(0);
      __syncthreads();                       // tile nb ready in LDS for all waves
      if (wave == 0 && (kc + KC) < K) {      // overlap: DMA next tile pair
        tdm_load_tile_2d((unsigned)(uintptr_t)&As[nb ^ 1][0][0],
                         &A[(size_t)m0 * K + kc + KC],
                         (unsigned)K, (unsigned)Mtot, KC, BM, (unsigned)K,
                         PAD_IVL_ENC, PAD_AMT_ENC);
        tdm_load_tile_2d((unsigned)(uintptr_t)&Ws[nb ^ 1][0][0],
                         &W[(size_t)n0 * K + kc + KC],
                         (unsigned)K, (unsigned)N, KC, BN, (unsigned)K,
                         PAD_IVL_ENC, PAD_AMT_ENC);
      }
#else
    int nb = 0;
    #pragma unroll 1
    for (int kc = 0; kc < K; kc += KC) {
      {
        int ra = tid >> 2, ca = (tid & 3) * 16;
        *(uint4*)&As[nb][ra][ca]     = *(const uint4*)&A[(size_t)(m0 + ra) * K + kc + ca];
        *(uint4*)&As[nb][ra][ca + 8] = *(const uint4*)&A[(size_t)(m0 + ra) * K + kc + ca + 8];
        int rw = tid >> 1, cw = (tid & 1) * 32;
        #pragma unroll
        for (int q = 0; q < 4; ++q)
          *(uint4*)&Ws[nb][rw][cw + q * 8] =
              *(const uint4*)&W[(size_t)(n0 + rw) * K + kc + cw + q * 8];
      }
      __syncthreads();
#endif
      // ---- 2 K-steps x (2x2) tiles = 8 WMMAs per wave per stage
      #pragma unroll
      for (int ks = 0; ks < 2; ++ks) {
        v16bf af[2], bfr[2];
        #pragma unroll
        for (int ti = 0; ti < 2; ++ti) {
          int ar = wm * 32 + ti * 16 + l16;
          af[ti] = ld_frag(&As[nb][ar][ks * 32 + hlf * 8],
                           &As[nb][ar][ks * 32 + 16 + hlf * 8]);
          int br = wn * 32 + ti * 16 + l16;
          bfr[ti] = ld_frag(&Ws[nb][br][ks * 32 + hlf * 8],
                            &Ws[nb][br][ks * 32 + 16 + hlf * 8]);
        }
        #pragma unroll
        for (int ti = 0; ti < 2; ++ti)
          #pragma unroll
          for (int tj = 0; tj < 2; ++tj)
            acc[ti][tj] = __builtin_amdgcn_wmma_f32_16x16x32_bf16(
                false, af[ti], false, bfr[tj], (short)0, acc[ti][tj], false, false);
      }
      nb ^= 1;
      __syncthreads();                       // done reading this buffer
    }
  }

  // ---- epilogue: + bias, store f32 (and optional bf16 mirror)
  #pragma unroll
  for (int ti = 0; ti < 2; ++ti) {
    #pragma unroll
    for (int tj = 0; tj < 2; ++tj) {
      int col = n0 + wn * 32 + tj * 16 + l16;
      float bsum = bias1[col];
      if (DUAL) bsum += bias2[col];
      #pragma unroll
      for (int r = 0; r < 8; ++r) {
        int row = m0 + wm * 32 + ti * 16 + r + 8 * hlf;
        float v = acc[ti][tj][r] + bsum;
        outF[(size_t)row * N + col] = v;
        if (OUTB) outB[(size_t)row * N + col] = (bf16)v;
      }
    }
  }
}

// ---------------- recurrent scan: h_t = tanh(h_{t-1} A^T + bx_t) ----------------
// 8 co-resident blocks; block owns 64 output columns with its A-slice in LDS.
// Full h state exchanged through L2 via hs[] + atomic grid barrier per step.
#define NBLK 8
#define GSL  64
#define LDAS 520   // 512 + 8 pad (1040B row stride: 16B aligned, bank-staggered)

__global__ __launch_bounds__(256) void k_scan(
    const float* __restrict__ Al,    // [H,H] row-major (rows = output g)
    const float* __restrict__ bx,    // [B,S,H] f32
    bf16* __restrict__ hs,           // [B,S,H] bf16 output
    unsigned* __restrict__ bar)
{
  __shared__ __align__(16) bf16 As[GSL][LDAS];
  __shared__ float red[4][32][8];

  const int tid  = threadIdx.x;
  const int lane = tid & 31;
  const int wave = tid >> 5;
  const int nt = wave & 3;     // N-tile (16 cols)
  const int kw = wave >> 2;    // K-half
  const int l16 = lane & 15, hlf = lane >> 4;
  const int g0 = blockIdx.x * GSL;

  // stage this block's 64x512 A-slice into LDS as bf16
  for (int i = tid; i < GSL * H_DIM; i += 256) {
    int r = i >> 9, c = i & 511;
    As[r][c] = (bf16)Al[(size_t)(g0 + r) * H_DIM + c];
  }
  __syncthreads();

  const int col = g0 + nt * 16 + l16;
  const v8f vzero = {};

  for (int t = 0; t < S_DIM; ++t) {
    v8f acc = vzero;
    if (kw == 0) {
      #pragma unroll
      for (int r = 0; r < 8; ++r) {
        int b = r + 8 * hlf;
        acc[r] = bx[((size_t)b * S_DIM + t) * H_DIM + col];
      }
    }
    if (t > 0) {
      const bf16* hrow = hs + ((size_t)l16 * S_DIM + (t - 1)) * H_DIM; // batch = l16
      #pragma unroll
      for (int kk = 0; kk < 8; ++kk) {
        int kc = (kw * 8 + kk) * 32;
        v16bf a = ld_frag(hrow + kc + hlf * 8, hrow + kc + 16 + hlf * 8);
        int br = nt * 16 + l16;
        v16bf b = ld_frag(&As[br][kc + hlf * 8], &As[br][kc + 16 + hlf * 8]);
        acc = __builtin_amdgcn_wmma_f32_16x16x32_bf16(
            false, a, false, b, (short)0, acc, false, false);
      }
    }

    __syncthreads();
    if (kw == 1) {
      #pragma unroll
      for (int r = 0; r < 8; ++r) red[nt][lane][r] = acc[r];
    }
    __syncthreads();
    if (kw == 0) {
      #pragma unroll
      for (int r = 0; r < 8; ++r) {
        float v = tanhf(acc[r] + red[nt][lane][r]);
        int b = r + 8 * hlf;
        hs[((size_t)b * S_DIM + t) * H_DIM + col] = (bf16)v;
      }
    }

    // device-wide step barrier
    __threadfence();
    __syncthreads();
    if (tid == 0) {
      atomicAdd(bar, 1u);
      unsigned target = (unsigned)NBLK * (unsigned)(t + 1);
      volatile unsigned* vb = bar;
      while (*vb < target) __builtin_amdgcn_s_sleep(1);
    }
    __syncthreads();
  }
}

__global__ void k_reset(unsigned* bar) { *bar = 0u; }

// ---------------- fused residual + LayerNorm ----------------
__global__ __launch_bounds__(256) void k_lnres(
    const float* __restrict__ xin, const float* __restrict__ y,
    const float* __restrict__ w, const float* __restrict__ b,
    float* __restrict__ xoutF, bf16* __restrict__ xoutB)
{
  __shared__ float ssum[8], ssq[8];
  const int row = blockIdx.x;
  const float* xr = xin + (size_t)row * H_DIM;
  const float* yr = y   + (size_t)row * H_DIM;
  const int c0 = threadIdx.x, c1 = threadIdx.x + 256;

  float v0 = xr[c0] + yr[c0];
  float v1 = xr[c1] + yr[c1];
  float s = v0 + v1, q = v0 * v0 + v1 * v1;
  #pragma unroll
  for (int off = 16; off; off >>= 1) {
    s += __shfl_down(s, off, 32);
    q += __shfl_down(q, off, 32);
  }
  const int lane = threadIdx.x & 31, wv = threadIdx.x >> 5;
  if (lane == 0) { ssum[wv] = s; ssq[wv] = q; }
  __syncthreads();
  if (wv == 0) {
    s = (lane < 8) ? ssum[lane] : 0.0f;
    q = (lane < 8) ? ssq[lane] : 0.0f;
    #pragma unroll
    for (int off = 4; off; off >>= 1) {
      s += __shfl_down(s, off, 32);
      q += __shfl_down(q, off, 32);
    }
    if (lane == 0) { ssum[0] = s; ssq[0] = q; }
  }
  __syncthreads();
  const float mu = ssum[0] * (1.0f / H_DIM);
  const float var = ssq[0] * (1.0f / H_DIM) - mu * mu;
  const float rs = rsqrtf(var + EPS_LN);

  float o0 = (v0 - mu) * rs * w[c0] + b[c0];
  float o1 = (v1 - mu) * rs * w[c1] + b[c1];
  xoutF[(size_t)row * H_DIM + c0] = o0;
  xoutF[(size_t)row * H_DIM + c1] = o1;
  xoutB[(size_t)row * H_DIM + c0] = (bf16)o0;
  xoutB[(size_t)row * H_DIM + c1] = (bf16)o1;
}

// ---------------- host orchestration ----------------
extern "C" void kernel_launch(void* const* d_in, const int* in_sizes, int n_in,
                              void* d_out, int out_size, void* d_ws, size_t ws_size,
                              hipStream_t stream)
{
  (void)in_sizes; (void)n_in; (void)out_size; (void)ws_size;

  const float* x     = (const float*)d_in[0];
  const float* in_W  = (const float*)d_in[1];
  const float* in_b  = (const float*)d_in[2];
  const float* A_W   = (const float*)d_in[3];
  const float* B_W   = (const float*)d_in[4];
  const float* B_b   = (const float*)d_in[5];
  const float* C_W   = (const float*)d_in[6];
  const float* C_b   = (const float*)d_in[7];
  const float* D_W   = (const float*)d_in[8];
  const float* D_b   = (const float*)d_in[9];
  const float* ln_w  = (const float*)d_in[10];
  const float* ln_b  = (const float*)d_in[11];
  const float* out_W = (const float*)d_in[12];
  const float* out_b = (const float*)d_in[13];

  char* ws = (char*)d_ws;
  size_t off = 0;
  auto alloc = [&](size_t bytes) -> char* {
    char* p = ws + off;
    off += (bytes + 255) & ~(size_t)255;
    return p;
  };
  bf16*  xb    = (bf16*) alloc((size_t)M_DIM * I_DIM * 2);
  bf16*  curb  = (bf16*) alloc((size_t)M_DIM * H_DIM * 2);
  bf16*  hsb   = (bf16*) alloc((size_t)M_DIM * H_DIM * 2);
  float* cur32 = (float*)alloc((size_t)M_DIM * H_DIM * 4);
  float* bx32  = (float*)alloc((size_t)M_DIM * H_DIM * 4);  // reused as y
  bf16*  inWb  = (bf16*) alloc((size_t)H_DIM * I_DIM * 2);
  bf16*  BWb   = (bf16*) alloc((size_t)L_DIM * H_DIM * H_DIM * 2);
  bf16*  CWb   = (bf16*) alloc((size_t)L_DIM * H_DIM * H_DIM * 2);
  bf16*  DWb   = (bf16*) alloc((size_t)L_DIM * H_DIM * H_DIM * 2);
  bf16*  outWb = (bf16*) alloc((size_t)O_DIM * H_DIM * 2);
  unsigned* bar = (unsigned*)alloc(256);

  // bf16 conversions
  k_cvt<<<1024, 256, 0, stream>>>(x,     xb,    M_DIM * I_DIM);
  k_cvt<<<256,  256, 0, stream>>>(in_W,  inWb,  H_DIM * I_DIM);
  k_cvt<<<1024, 256, 0, stream>>>(B_W,   BWb,   L_DIM * H_DIM * H_DIM);
  k_cvt<<<1024, 256, 0, stream>>>(C_W,   CWb,   L_DIM * H_DIM * H_DIM);
  k_cvt<<<1024, 256, 0, stream>>>(D_W,   DWb,   L_DIM * H_DIM * H_DIM);
  k_cvt<<<256,  256, 0, stream>>>(out_W, outWb, O_DIM * H_DIM);

  dim3 gH(H_DIM / BN, M_DIM / BM);  // (4, 512)
  dim3 gO(O_DIM / BN, M_DIM / BM);  // (2, 512)

  // input projection -> cur (f32 + bf16)
  k_gemm<false, true><<<gH, 256, 0, stream>>>(
      xb, inWb, in_b, I_DIM, nullptr, nullptr, nullptr, 0,
      cur32, curb, H_DIM, M_DIM);

  for (int l = 0; l < L_DIM; ++l) {
    const size_t wo = (size_t)l * H_DIM * H_DIM;
    // bx = cur @ B^T + B_b
    k_gemm<false, false><<<gH, 256, 0, stream>>>(
        curb, BWb + wo, B_b + l * H_DIM, H_DIM, nullptr, nullptr, nullptr, 0,
        bx32, nullptr, H_DIM, M_DIM);
    // recurrent scan -> hs (bf16)
    k_reset<<<1, 1, 0, stream>>>(bar);
    k_scan<<<NBLK, 256, 0, stream>>>(A_W + wo, bx32, hsb, bar);
    // y = hs @ C^T + cur @ D^T + C_b + D_b   (dual-K fused GEMM, into bx32)
    k_gemm<true, false><<<gH, 256, 0, stream>>>(
        hsb, CWb + wo, C_b + l * H_DIM, H_DIM,
        curb, DWb + wo, D_b + l * H_DIM, H_DIM,
        bx32, nullptr, H_DIM, M_DIM);
    // cur = LN(cur + y)
    k_lnres<<<M_DIM, 256, 0, stream>>>(cur32, bx32,
                                       ln_w + l * H_DIM, ln_b + l * H_DIM,
                                       cur32, curb);
  }

  // output projection -> d_out (f32)
  k_gemm<false, false><<<gO, 256, 0, stream>>>(
      curb, outWb, out_b, H_DIM, nullptr, nullptr, nullptr, 0,
      (float*)d_out, nullptr, O_DIM, M_DIM);
}
